// EdgeGCNConv_32701880992041
// MI455X (gfx1250) — matrix operand order repaired and checked by reading, not compiled, not run on hardware
//
#include <hip/hip_runtime.h>
#include <hip/hip_bf16.h>

typedef __attribute__((ext_vector_type(16))) _Float16 v16h;
typedef __attribute__((ext_vector_type(8)))  _Float16 v8h;
typedef __attribute__((ext_vector_type(4)))  _Float16 v4h;
typedef __attribute__((ext_vector_type(8)))  float    v8f;
typedef __attribute__((ext_vector_type(4)))  float    v4f;

#define CN      128          // node feature width
#define CE      64           // edge feature width
#define COUT    128          // output channels
#define KTOT    320          // 2*CN + CE
#define NSTEPS  10           // KTOT / 32 (WMMA K-steps)
#define KPAD    392          // padded A-row length in f16 (196 DWORDs; 196 % 64 == 4)
#define TILE_E  128          // edges per tile
#define MAXBLK  2048         // persistent-block cap (amortizes weight fetch)

// ---------------------------------------------------------------------------
// Prep: pack [W_pass | W_self] -> f16 Wc[128][320], bias = b_pass + b_self
// ---------------------------------------------------------------------------
__global__ void edgegcn_prep(const float* __restrict__ Wp, const float* __restrict__ bp,
                             const float* __restrict__ Ws, const float* __restrict__ bs,
                             _Float16* __restrict__ Wc, float* __restrict__ bias) {
  int i = blockIdx.x * blockDim.x + threadIdx.x;
  if (i < COUT * KTOT) {
    int o = i / KTOT;
    int k = i - o * KTOT;
    float v = (k < 2 * CN) ? Wp[o * (2 * CN) + k] : Ws[o * CE + (k - 2 * CN)];
    Wc[i] = (_Float16)v;
  }
  if (i < COUT) bias[i] = bp[i] + bs[i];
}

// ---------------------------------------------------------------------------
// Main: persistent blocks; per tile: gather -> f16 A-tile in LDS ->
//       v_wmma_f32_16x16x32_f16 (B in registers) -> bias + ReLU -> NT stores
// ---------------------------------------------------------------------------
__global__ __launch_bounds__(256) void edgegcn_main(
    const float* __restrict__ X, const float* __restrict__ EV,
    const long long* __restrict__ EIdx, const _Float16* __restrict__ Wc,
    const float* __restrict__ bias, float* __restrict__ out, int E, int ntiles) {
  extern __shared__ _Float16 sA[];               // [TILE_E][KPAD] f16, ~100 KB

  const int tid  = threadIdx.x;
  const int lane = tid & 31;
  const int wave = tid >> 5;                     // n-tile (output channel slab) 0..7
  const int lmod = lane & 15;
  const int lhi  = lane >> 4;                    // half-wave select

  // --- B fragments, loaded ONCE per persistent block.
  //     Contiguous-k layout used here (matches A) for 16-bit B 32x16: lanes 0-15 hold K=0..15 of column
  //     N=lane, lanes 16-31 hold K=16..31 -> one contiguous 32B read per step.
  v16h bfrag[NSTEPS];
  {
    const _Float16* wrow = Wc + (wave * 16 + lmod) * KTOT + lhi * 16;
#pragma unroll
    for (int ks = 0; ks < NSTEPS; ++ks)
      bfrag[ks] = *(const v16h*)(wrow + ks * 32);
  }
  const float bias_n = bias[wave * 16 + lmod];

  // Phase-A thread mapping: 2 threads per edge
  const int ea   = tid >> 1;                     // edge slot in tile [0,128)
  const int half = tid & 1;                      // channel half

  for (int t = blockIdx.x; t < ntiles; t += gridDim.x) {
    const long long e0 = (long long)t * TILE_E;

    // --- Cooperative A-tile build: gather, combine, cvt f16, stage in LDS ---
    {
      long long eg = e0 + ea;
      if (eg >= E) eg = E - 1;                   // clamp (tail safety)
      const long long src = __builtin_nontemporal_load(EIdx + eg);
      const long long dst = __builtin_nontemporal_load(EIdx + (long long)E + eg);
      // X gathers: keep regular-temporal -> X (51 MB) stays L2-resident
      const v4f* __restrict__ x1 = (const v4f*)(X + src * CN) + half * 16;
      const v4f* __restrict__ x2 = (const v4f*)(X + dst * CN) + half * 16;
      _Float16* rowA = sA + ea * KPAD;
#pragma unroll
      for (int j = 0; j < 16; ++j) {             // 64 channels of d and s
        v4f a = x1[j], b = x2[j];
        int c = half * 64 + j * 4;
        v4h d, s;
        d[0] = (_Float16)((a[0] - b[0]) * 0.5f); d[1] = (_Float16)((a[1] - b[1]) * 0.5f);
        d[2] = (_Float16)((a[2] - b[2]) * 0.5f); d[3] = (_Float16)((a[3] - b[3]) * 0.5f);
        s[0] = (_Float16)((a[0] + b[0]) * 0.5f); s[1] = (_Float16)((a[1] + b[1]) * 0.5f);
        s[2] = (_Float16)((a[2] + b[2]) * 0.5f); s[3] = (_Float16)((a[3] + b[3]) * 0.5f);
        *(v4h*)(rowA + c)      = d;              // K = [0,128): d
        *(v4h*)(rowA + CN + c) = s;              // K = [128,256): s
      }
      // edge_values: streamed once -> non-temporal loads (don't pollute L2)
      const v4f* __restrict__ ev = (const v4f*)(EV + eg * CE) + half * 8;
#pragma unroll
      for (int j = 0; j < 8; ++j) {              // 32 channels of edge_values
        v4f a = __builtin_nontemporal_load(ev + j);
        int c = half * 32 + j * 4;
        v4h h;
        h[0] = (_Float16)a[0]; h[1] = (_Float16)a[1];
        h[2] = (_Float16)a[2]; h[3] = (_Float16)a[3];
        *(v4h*)(rowA + 2 * CN + c) = h;          // K = [256,320): edge_values
      }
    }
    __syncthreads();

    // --- WMMA: wave owns channels [wave*16, wave*16+16), loops 8 edge tiles.
    //     16-bit A 16x32 layout: lane<16 (M=lane) holds K {0..15},
    //     lane>=16 (M=lane-16) holds K {16..31} -> two 16B LDS reads.
    const bool full = (e0 + TILE_E) <= (long long)E;   // block-uniform guard
#pragma unroll 1
    for (int mt = 0; mt < 8; ++mt) {
      v8f acc = {};
      const _Float16* arow = sA + (mt * 16 + lmod) * KPAD + lhi * 16;
#pragma unroll
      for (int ks = 0; ks < NSTEPS; ++ks) {
        v8h a0 = *(const v8h*)(arow + ks * 32);
        v8h a1 = *(const v8h*)(arow + ks * 32 + 8);
        v16h a = __builtin_shufflevector(a0, a1, 0, 1, 2, 3, 4, 5, 6, 7,
                                         8, 9, 10, 11, 12, 13, 14, 15);
        acc = __builtin_amdgcn_wmma_f32_16x16x32_f16(false, a, false, bfrag[ks],
                                                     (short)0, acc, false, false);
      }
      // D layout: VGPR r -> row M = mt*16 + r + 8*lhi, lane%16 -> channel N
      const long long mrow = e0 + mt * 16 + lhi * 8;
      float* __restrict__ obase = out + mrow * COUT + wave * 16 + lmod;
      if (full) {
#pragma unroll
        for (int r = 0; r < 8; ++r) {
          float v = fmaxf(acc[r] + bias_n, 0.0f);
          __builtin_nontemporal_store(v, obase + (long long)r * COUT);
        }
      } else {
#pragma unroll
        for (int r = 0; r < 8; ++r) {
          if (mrow + r < (long long)E) {
            float v = fmaxf(acc[r] + bias_n, 0.0f);
            __builtin_nontemporal_store(v, obase + (long long)r * COUT);
          }
        }
      }
    }
    __syncthreads();   // protect sA before next tile's writes
  }
}

// ---------------------------------------------------------------------------
extern "C" void kernel_launch(void* const* d_in, const int* in_sizes, int n_in,
                              void* d_out, int out_size, void* d_ws, size_t ws_size,
                              hipStream_t stream) {
  const float*     X    = (const float*)d_in[0];
  const float*     EV   = (const float*)d_in[1];
  const float*     Wp   = (const float*)d_in[2];
  const float*     bp   = (const float*)d_in[3];
  const float*     Ws   = (const float*)d_in[4];
  const float*     bs   = (const float*)d_in[5];
  const long long* EIdx = (const long long*)d_in[6];
  float*           out  = (float*)d_out;
  const int E = in_sizes[6] / 2;

  // workspace: f16 combined weights (81,920 B) + combined bias (512 B)
  _Float16* Wc  = (_Float16*)d_ws;
  float*    bia = (float*)((char*)d_ws + (size_t)COUT * KTOT * sizeof(_Float16));

  edgegcn_prep<<<(COUT * KTOT + 255) / 256, 256, 0, stream>>>(Wp, bp, Ws, bs, Wc, bia);

  const size_t lds_bytes = (size_t)TILE_E * KPAD * sizeof(_Float16);  // 100,352 B
  // idempotent, immediate (not a stream op): allow >64KB dynamic LDS
  (void)hipFuncSetAttribute((const void*)edgegcn_main,
                            hipFuncAttributeMaxDynamicSharedMemorySize, (int)lds_bytes);

  const int ntiles = (E + TILE_E - 1) / TILE_E;
  const int nblk   = ntiles < MAXBLK ? ntiles : MAXBLK;
  edgegcn_main<<<nblk, 256, lds_bytes, stream>>>(X, EV, EIdx, Wc, bia, out, E, ntiles);
}
